// EfficientSABlock_3822520893838
// MI455X (gfx1250) — matrix operand order repaired and checked by reading, not compiled, not run on hardware
//
#include <hip/hip_runtime.h>
#include <hip/hip_bf16.h>

typedef __attribute__((ext_vector_type(16))) __bf16 v16bf;
typedef __attribute__((ext_vector_type(8)))  __bf16 bf8;
typedef __attribute__((ext_vector_type(8)))  float  v8f;
typedef __attribute__((ext_vector_type(4)))  int    v4i;

// ---------------------------------------------------------------------------
// CDNA5 async global->LDS staging (ASYNCcnt path). Guarded so the file still
// compiles on toolchains that do not declare the builtin.
// Probe result (round 2): param 0 is 'v4i addrspace(1)*' (global, typed),
// param 1 is the LDS pointer, then imm offset + imm cpol.
// ---------------------------------------------------------------------------
#if defined(__HIP_DEVICE_COMPILE__) && \
    __has_builtin(__builtin_amdgcn_global_load_async_to_lds_b128) && \
    __has_builtin(__builtin_amdgcn_s_wait_asynccnt)
#define HAVE_ASYNC_LDS 1
#endif

#ifdef HAVE_ASYNC_LDS
__device__ __forceinline__ void async_cp16(const void* g, void* l) {
    __builtin_amdgcn_global_load_async_to_lds_b128(
        (__attribute__((address_space(1))) v4i*)(g),
        (__attribute__((address_space(3))) v4i*)(l), 0, 0);
}
__device__ __forceinline__ void async_wait0() {
    __builtin_amdgcn_s_wait_asynccnt(0);
}
#endif

// ---------------------------------------------------------------------------
// WMMA fragment loaders (bf16, 16x16x32). LDS tiles row-major:
//   A tile [m][k]; B tile stored transposed as [n][k].
// Per ISA 7.12.2:
//   A frag lane L (m=L%16, half=L/16): j=0..7  -> k = koff + half*8 + j
//                                      j=8..15 -> k = koff + 16 + half*8 + (j-8)
//   B frag lane L (n=L%16, half=L/16): same k mapping as A frag
// Row strides are multiples of 8 bf16 (16B) so all bf8 loads are aligned.
// ---------------------------------------------------------------------------
__device__ __forceinline__ v16bf load_a_frag(const __bf16* base, int stride,
                                             int m, int koff, int hf) {
    const __bf16* p = base + (size_t)m * stride + koff + hf * 8;
    union { v16bf v; bf8 h[2]; } u;
    u.h[0] = *(const bf8*)(p);
    u.h[1] = *(const bf8*)(p + 16);
    return u.v;
}

__device__ __forceinline__ v16bf load_b_frag(const __bf16* base, int stride,
                                             int n, int koff, int hf) {
    const __bf16* p = base + (size_t)n * stride + koff + hf * 8;
    union { v16bf v; bf8 h[2]; } u;
    u.h[0] = *(const bf8*)(p);
    u.h[1] = *(const bf8*)(p + 16);
    return u.v;
}

__device__ __forceinline__ v8f wmma_bf16(v16bf a, v16bf b, v8f c) {
    return __builtin_amdgcn_wmma_f32_16x16x32_bf16(false, a, false, b,
                                                   (short)0, c, false, false);
}

// ---------------------------------------------------------------------------
// GEMM tiling (k1/k3/k4/k6): block = 64(M) x 128(N), K chunked by 128.
// 8 waves: wm = w&3 (row group of 16), wn = w>>2 (col group of 64).
// Each wave: 1 A frag reused across 4 B frags -> 4 WMMA per k-step.
// LDS: A 64x136 bf16 (17.0KB) + B 128x136 bf16 (34.0KB) = 51KB.
// ---------------------------------------------------------------------------

// Kernel 1: spatial-reduction conv as GEMM.
//   M=2048 (b*256 pixels), N=256 (c_out), K=4096 (c_in*16), 32 K-chunks.
//   A[row][k] = x[b, (4i+kh)*64 + 4j+kw, c_in]  (im2col on the fly)
//   B[k][n]   = sr_w[n][k]  (sr_w is [Cout][Cin*4*4] row-major already)
__global__ __launch_bounds__(256)
void k1_conv_gemm(const float* __restrict__ x, const float* __restrict__ srw,
                  const float* __restrict__ srb, float* __restrict__ xkv) {
    __shared__ __bf16 Al[64][136];
    __shared__ __bf16 Bl[128][136];
    const int tid = threadIdx.x;
    const int w = tid >> 5, lane = tid & 31, hf = lane >> 4, l16 = lane & 15;
    const int wm = w & 3, wn = w >> 2;
    const int m0 = blockIdx.x * 64, n0 = blockIdx.y * 128;
    v8f acc[4] = {};
    for (int kc = 0; kc < 32; ++kc) {
        const int k0 = kc * 128;
        for (int e = tid; e < 64 * 128; e += 256) {
            const int r = e >> 7, kk = e & 127;
            const int k = k0 + kk;
            const int cin = k >> 4, kh = (k >> 2) & 3, kw = k & 3;
            const int row = m0 + r;
            const int b = row >> 8, pix = row & 255;
            const int pi = pix >> 4, pj = pix & 15;
            const int ntok = (4 * pi + kh) * 64 + (4 * pj + kw);
            Al[r][kk] = (__bf16)x[((size_t)b * 4096 + ntok) * 256 + cin];
        }
        for (int e = tid; e < 128 * 128; e += 256) {
            const int kk = e & 127, n = e >> 7;   // lanes contiguous in k
            Bl[n][kk] = (__bf16)srw[(size_t)(n0 + n) * 4096 + k0 + kk];
        }
        __syncthreads();
#pragma unroll
        for (int ks = 0; ks < 4; ++ks) {
            v16bf af = load_a_frag(&Al[0][0], 136, wm * 16 + l16, ks * 32, hf);
#pragma unroll
            for (int t = 0; t < 4; ++t) {
                v16bf bfr = load_b_frag(&Bl[0][0], 136, wn * 64 + t * 16 + l16,
                                        ks * 32, hf);
                acc[t] = wmma_bf16(af, bfr, acc[t]);
            }
        }
        __syncthreads();
    }
#pragma unroll
    for (int t = 0; t < 4; ++t) {
        const int col = n0 + wn * 64 + t * 16 + l16;
        const float bias = srb[col];
#pragma unroll
        for (int r = 0; r < 8; ++r) {
            const int m = m0 + wm * 16 + r + hf * 8;
            xkv[(size_t)m * 256 + col] = acc[t][r] + bias;
        }
    }
}

// Kernel 2: LayerNorm over C=256 for 2048 reduced tokens; bf16 out.
__global__ __launch_bounds__(256)
void k2_layernorm(const float* __restrict__ xkv, const float* __restrict__ g,
                  const float* __restrict__ bta, __bf16* __restrict__ xln) {
    __shared__ float red[256];
    const int r = blockIdx.x, t = threadIdx.x;
    const float v = xkv[(size_t)r * 256 + t];
    red[t] = v;
    __syncthreads();
    for (int s = 128; s > 0; s >>= 1) {
        if (t < s) red[t] += red[t + s];
        __syncthreads();
    }
    const float mu = red[0] * (1.0f / 256.0f);
    __syncthreads();
    const float dv = v - mu;
    red[t] = dv * dv;
    __syncthreads();
    for (int s = 128; s > 0; s >>= 1) {
        if (t < s) red[t] += red[t + s];
        __syncthreads();
    }
    const float var = red[0] * (1.0f / 256.0f);
    const float y = dv * rsqrtf(var + 1e-5f) * g[t] + bta[t];
    xln[(size_t)r * 256 + t] = (__bf16)y;
}

// Kernel 3: KV projection. M=2048, N=512, K=256 (2 chunks).
//   Output scattered to k_bf/v_bf [b][h][nk][d]  (col = two*256 + h*64 + d).
__global__ __launch_bounds__(256)
void k3_kv_gemm(const __bf16* __restrict__ xln, const float* __restrict__ wkv,
                __bf16* __restrict__ kb, __bf16* __restrict__ vb) {
    __shared__ __bf16 Al[64][136];
    __shared__ __bf16 Bl[128][136];
    const int tid = threadIdx.x;
    const int w = tid >> 5, lane = tid & 31, hf = lane >> 4, l16 = lane & 15;
    const int wm = w & 3, wn = w >> 2;
    const int m0 = blockIdx.x * 64, n0 = blockIdx.y * 128;
    v8f acc[4] = {};
    for (int kc = 0; kc < 2; ++kc) {
        const int k0 = kc * 128;
#ifdef HAVE_ASYNC_LDS
        for (int e = tid; e < 64 * 16; e += 256) {     // 16B granules
            const int r = e >> 4, part = e & 15;
            async_cp16(xln + (size_t)(m0 + r) * 256 + k0 + part * 8,
                       &Al[r][part * 8]);
        }
#else
        for (int e = tid; e < 64 * 128; e += 256) {
            const int r = e >> 7, kk = e & 127;
            Al[r][kk] = xln[(size_t)(m0 + r) * 256 + k0 + kk];
        }
#endif
        for (int e = tid; e < 128 * 128; e += 256) {
            const int n = e & 127, kk = e >> 7;   // lanes contiguous in n
            Bl[n][kk] = (__bf16)wkv[(size_t)(k0 + kk) * 512 + n0 + n];
        }
#ifdef HAVE_ASYNC_LDS
        async_wait0();
#endif
        __syncthreads();
#pragma unroll
        for (int ks = 0; ks < 4; ++ks) {
            v16bf af = load_a_frag(&Al[0][0], 136, wm * 16 + l16, ks * 32, hf);
#pragma unroll
            for (int t = 0; t < 4; ++t) {
                v16bf bfr = load_b_frag(&Bl[0][0], 136, wn * 64 + t * 16 + l16,
                                        ks * 32, hf);
                acc[t] = wmma_bf16(af, bfr, acc[t]);
            }
        }
        __syncthreads();
    }
#pragma unroll
    for (int t = 0; t < 4; ++t) {
        const int cg = n0 + wn * 64 + t * 16 + l16;
        const int two = cg >> 8, hh = (cg >> 6) & 3, d = cg & 63;
        __bf16* dst = two ? vb : kb;
#pragma unroll
        for (int r = 0; r < 8; ++r) {
            const int rg = m0 + wm * 16 + r + hf * 8;
            const int b = rg >> 8, nk = rg & 255;
            dst[(((size_t)b * 4 + hh) * 256 + nk) * 64 + d] = (__bf16)acc[t][r];
        }
    }
}

// Kernel 4: Q projection. M=32768, N=256, K=256.  Out: q_bf [b][h][n][d].
__global__ __launch_bounds__(256)
void k4_q_gemm(const float* __restrict__ x, const float* __restrict__ wq,
               __bf16* __restrict__ qb) {
    __shared__ __bf16 Al[64][136];
    __shared__ __bf16 Bl[128][136];
    const int tid = threadIdx.x;
    const int w = tid >> 5, lane = tid & 31, hf = lane >> 4, l16 = lane & 15;
    const int wm = w & 3, wn = w >> 2;
    const int m0 = blockIdx.x * 64, n0 = blockIdx.y * 128;
    v8f acc[4] = {};
    for (int kc = 0; kc < 2; ++kc) {
        const int k0 = kc * 128;
        for (int e = tid; e < 64 * 128; e += 256) {
            const int r = e >> 7, kk = e & 127;
            Al[r][kk] = (__bf16)x[(size_t)(m0 + r) * 256 + k0 + kk];
        }
        for (int e = tid; e < 128 * 128; e += 256) {
            const int n = e & 127, kk = e >> 7;
            Bl[n][kk] = (__bf16)wq[(size_t)(k0 + kk) * 256 + n0 + n];
        }
        __syncthreads();
#pragma unroll
        for (int ks = 0; ks < 4; ++ks) {
            v16bf af = load_a_frag(&Al[0][0], 136, wm * 16 + l16, ks * 32, hf);
#pragma unroll
            for (int t = 0; t < 4; ++t) {
                v16bf bfr = load_b_frag(&Bl[0][0], 136, wn * 64 + t * 16 + l16,
                                        ks * 32, hf);
                acc[t] = wmma_bf16(af, bfr, acc[t]);
            }
        }
        __syncthreads();
    }
#pragma unroll
    for (int t = 0; t < 4; ++t) {
        const int cg = n0 + wn * 64 + t * 16 + l16;
        const int hh = cg >> 6, d = cg & 63;
#pragma unroll
        for (int r = 0; r < 8; ++r) {
            const int rg = m0 + wm * 16 + r + hf * 8;
            const int b = rg >> 12, nt = rg & 4095;
            qb[(((size_t)b * 4 + hh) * 4096 + nt) * 64 + d] = (__bf16)acc[t][r];
        }
    }
}

// ---------------------------------------------------------------------------
// Kernel 5: attention per (b,h). 32 queries/block vs all 256 keys.
//   S = Q K^T * 0.125 -> exact row softmax -> O = P V
// LDS plan (static 53760B):
//   regA [0,16896):   Q[32][72]   then P[32][264]
//   regB [16896,..):  K[256][72]  then S f32[32][260] then V^T[64][264]
// ---------------------------------------------------------------------------
__global__ __launch_bounds__(256)
void k5_attention(const __bf16* __restrict__ q, const __bf16* __restrict__ kbuf,
                  const __bf16* __restrict__ vbuf, __bf16* __restrict__ attout) {
    __shared__ __align__(16) char smem[16896 + 36864];
    __bf16* Ql = (__bf16*)smem;             // [32][72]
    __bf16* Pl = (__bf16*)smem;             // [32][264]
    __bf16* Kl = (__bf16*)(smem + 16896);   // [256][72]
    float*  Sl = (float*)(smem + 16896);    // [32][260]
    __bf16* Vt = (__bf16*)(smem + 16896);   // [64][264]
    const int tid = threadIdx.x;
    const int w = tid >> 5, lane = tid & 31, hf = lane >> 4, l16 = lane & 15;
    const int bh = blockIdx.y;
    const int mblk = blockIdx.x * 32;
    const __bf16* qh = q + (size_t)bh * 4096 * 64;
    const __bf16* kh = kbuf + (size_t)bh * 256 * 64;
    const __bf16* vh = vbuf + (size_t)bh * 256 * 64;

#ifdef HAVE_ASYNC_LDS
    {
        const int r = tid >> 3, part = tid & 7;          // Q: 32x8 granules
        async_cp16(qh + (size_t)(mblk + r) * 64 + part * 8, Ql + r * 72 + part * 8);
    }
    for (int e = tid; e < 256 * 8; e += 256) {           // K: 256x8 granules
        const int r = e >> 3, part = e & 7;
        async_cp16(kh + (size_t)r * 64 + part * 8, Kl + r * 72 + part * 8);
    }
    async_wait0();
#else
    for (int e = tid; e < 32 * 64; e += 256) {
        const int r = e >> 6, d = e & 63;
        Ql[r * 72 + d] = qh[(size_t)(mblk + r) * 64 + d];
    }
    for (int e = tid; e < 256 * 64; e += 256) {
        const int nk = e >> 6, d = e & 63;
        Kl[nk * 72 + d] = kh[e];
    }
#endif
    __syncthreads();

    // ---- S = Q K^T ----
    const int wmr = w & 1, wc = w >> 1;   // wave: 16 rows x 64 cols strip
    v8f sacc[4] = {};
#pragma unroll
    for (int ks = 0; ks < 2; ++ks) {
        v16bf af = load_a_frag(Ql, 72, wmr * 16 + l16, ks * 32, hf);
#pragma unroll
        for (int t = 0; t < 4; ++t) {
            v16bf bfr = load_b_frag(Kl, 72, wc * 64 + t * 16 + l16, ks * 32, hf);
            sacc[t] = wmma_bf16(af, bfr, sacc[t]);
        }
    }
    __syncthreads();   // all K reads complete before S overlays the region
#pragma unroll
    for (int t = 0; t < 4; ++t) {
        const int col = wc * 64 + t * 16 + l16;
#pragma unroll
        for (int r = 0; r < 8; ++r) {
            const int m = wmr * 16 + r + hf * 8;
            Sl[m * 260 + col] = sacc[t][r] * 0.125f;  // hd^-0.5
        }
    }
    __syncthreads();

    // ---- exact row softmax: wave w handles rows 4w..4w+3 ----
    for (int rr = 0; rr < 4; ++rr) {
        const int m = w * 4 + rr;
        float sv[8];
        float vmax = -1e30f;
#pragma unroll
        for (int j = 0; j < 8; ++j) {
            sv[j] = Sl[m * 260 + lane + j * 32];
            vmax = fmaxf(vmax, sv[j]);
        }
#pragma unroll
        for (int off = 16; off > 0; off >>= 1)
            vmax = fmaxf(vmax, __shfl_xor(vmax, off, 32));
        float sum = 0.f;
#pragma unroll
        for (int j = 0; j < 8; ++j) {
            sv[j] = __expf(sv[j] - vmax);
            sum += sv[j];
        }
#pragma unroll
        for (int off = 16; off > 0; off >>= 1)
            sum += __shfl_xor(sum, off, 32);
        const float inv = 1.0f / sum;
#pragma unroll
        for (int j = 0; j < 8; ++j)
            Pl[m * 264 + lane + j * 32] = (__bf16)(sv[j] * inv);
    }
    __syncthreads();

    // ---- stage V transposed (overlays S) ----
    for (int e = tid; e < 256 * 64; e += 256) {
        const int nk = e >> 6, d = e & 63;
        Vt[d * 264 + nk] = vh[e];
    }
    __syncthreads();

    // ---- O = P V ----
    const int wm2 = w >> 2, wn2 = w & 3;
    v8f oacc = {};
#pragma unroll
    for (int ks = 0; ks < 8; ++ks) {
        v16bf af = load_a_frag(Pl, 264, wm2 * 16 + l16, ks * 32, hf);
        v16bf bfr = load_b_frag(Vt, 264, wn2 * 16 + l16, ks * 32, hf);
        oacc = wmma_bf16(af, bfr, oacc);
    }
    const int b = bh >> 2, hh = bh & 3;
    const int d = wn2 * 16 + l16;
#pragma unroll
    for (int r = 0; r < 8; ++r) {
        const int m = mblk + wm2 * 16 + r + hf * 8;
        attout[((size_t)b * 4096 + m) * 256 + hh * 64 + d] = (__bf16)oacc[r];
    }
}

// Kernel 6: output projection. M=32768, N=256, K=256, + bias, fp32 out.
__global__ __launch_bounds__(256)
void k6_out_gemm(const __bf16* __restrict__ ain, const float* __restrict__ wo,
                 const float* __restrict__ bo, float* __restrict__ out) {
    __shared__ __bf16 Al[64][136];
    __shared__ __bf16 Bl[128][136];
    const int tid = threadIdx.x;
    const int w = tid >> 5, lane = tid & 31, hf = lane >> 4, l16 = lane & 15;
    const int wm = w & 3, wn = w >> 2;
    const int m0 = blockIdx.x * 64, n0 = blockIdx.y * 128;
    v8f acc[4] = {};
    for (int kc = 0; kc < 2; ++kc) {
        const int k0 = kc * 128;
#ifdef HAVE_ASYNC_LDS
        for (int e = tid; e < 64 * 16; e += 256) {
            const int r = e >> 4, part = e & 15;
            async_cp16(ain + (size_t)(m0 + r) * 256 + k0 + part * 8,
                       &Al[r][part * 8]);
        }
#else
        for (int e = tid; e < 64 * 128; e += 256) {
            const int r = e >> 7, kk = e & 127;
            Al[r][kk] = ain[(size_t)(m0 + r) * 256 + k0 + kk];
        }
#endif
        for (int e = tid; e < 128 * 128; e += 256) {
            const int n = e & 127, kk = e >> 7;
            Bl[n][kk] = (__bf16)wo[(size_t)(k0 + kk) * 256 + n0 + n];
        }
#ifdef HAVE_ASYNC_LDS
        async_wait0();
#endif
        __syncthreads();
#pragma unroll
        for (int ks = 0; ks < 4; ++ks) {
            v16bf af = load_a_frag(&Al[0][0], 136, wm * 16 + l16, ks * 32, hf);
#pragma unroll
            for (int t = 0; t < 4; ++t) {
                v16bf bfr = load_b_frag(&Bl[0][0], 136, wn * 64 + t * 16 + l16,
                                        ks * 32, hf);
                acc[t] = wmma_bf16(af, bfr, acc[t]);
            }
        }
        __syncthreads();
    }
#pragma unroll
    for (int t = 0; t < 4; ++t) {
        const int cg = n0 + wn * 64 + t * 16 + l16;
        const float bias = bo[cg];
#pragma unroll
        for (int r = 0; r < 8; ++r) {
            const int rg = m0 + wm * 16 + r + hf * 8;
            out[(size_t)rg * 256 + cg] = acc[t][r] + bias;
        }
    }
}

// ---------------------------------------------------------------------------
// Host launcher
// ---------------------------------------------------------------------------
extern "C" void kernel_launch(void* const* d_in, const int* in_sizes, int n_in,
                              void* d_out, int out_size, void* d_ws, size_t ws_size,
                              hipStream_t stream) {
    const float* x    = (const float*)d_in[0];
    const float* Wq   = (const float*)d_in[3];
    const float* Wkv  = (const float*)d_in[4];
    const float* sr_w = (const float*)d_in[5];
    const float* sr_b = (const float*)d_in[6];
    const float* ln_g = (const float*)d_in[7];
    const float* ln_b = (const float*)d_in[8];
    const float* Wo   = (const float*)d_in[9];
    const float* bo   = (const float*)d_in[10];
    float* out = (float*)d_out;

    // workspace layout (bytes), total 38,797,312
    char* ws = (char*)d_ws;
    __bf16* qb  = (__bf16*)(ws);                 // 16,777,216  Q bf16 [b][h][n][d]
    float*  xkv = (float*) (ws + 16777216);      //  2,097,152  conv out fp32
    __bf16* xln = (__bf16*)(ws + 18874368);      //  1,048,576  LN out bf16
    __bf16* kb  = (__bf16*)(ws + 19922944);      //  1,048,576  K bf16
    __bf16* vb  = (__bf16*)(ws + 20971520);      //  1,048,576  V bf16
    __bf16* ao  = (__bf16*)(ws + 22020096);      // 16,777,216  attn out bf16

    k1_conv_gemm<<<dim3(32, 2), 256, 0, stream>>>(x, sr_w, sr_b, xkv);
    k2_layernorm<<<dim3(2048), 256, 0, stream>>>(xkv, ln_g, ln_b, xln);
    k3_kv_gemm<<<dim3(32, 4), 256, 0, stream>>>(xln, Wkv, kb, vb);
    k4_q_gemm<<<dim3(512, 2), 256, 0, stream>>>(x, Wq, qb);
    k5_attention<<<dim3(128, 32), 256, 0, stream>>>(qb, kb, vb, ao);
    k6_out_gemm<<<dim3(512, 2), 256, 0, stream>>>(ao, Wo, bo, out);
}
